// CausalTransitionModel_86303072845885
// MI455X (gfx1250) — matrix operand order repaired and checked by reading, not compiled, not run on hardware
//
#include <hip/hip_runtime.h>

// ---------------------------------------------------------------------------
// Fused CausalTransitionModel for MI455X (gfx1250), wave32 + bf16 WMMA.
// Pass 1: transpose+convert all weights to bf16 in d_ws (B-frag = 2x b128).
// Pass 2: one workgroup per batch element; all intermediates in LDS.
// ---------------------------------------------------------------------------

typedef __attribute__((ext_vector_type(16))) __bf16 v16bf;
typedef __attribute__((ext_vector_type(8)))  __bf16 v8bf;
typedef __attribute__((ext_vector_type(8)))  float  v8f;

#define KOBJ 16
#define DDIM 128
#define HDIM 512
#define ADIM 8
#define CH   48            // edge rows per chunk (240 = 5 * 48, 3 M-tiles)
#define NCH  5
#define LDSTR 528          // bf16 row stride (512 + 16 pad, 16B aligned)
#define NAK   672          // node GEMM1 K padded: 128+8+512=648 -> 672
#define NASTR 688          // node-A LDS row stride (16B aligned)
#define NTHREADS 256
#define LN_EPS 1e-5f

// d_ws layout (bf16 elements), all 16-element aligned
#define EW1T_OFF 0u               // [512][256]
#define EW2T_OFF 131072u          // [512][512]
#define EW3T_OFF 393216u          // [512][512]
#define NW1T_OFF 655360u          // [512][672] (zero-padded k>=648)
#define NW2T_OFF 999424u          // [512][512]
#define NW3T_OFF 1261568u         // [128][512]
#define WS_TOTAL 1327104u

// ---- fragment loaders (contiguous k = 16h + s, same in A and B) --------------------------------
// A 16x32 bf16: lane holds row m=lane&15; elements are two contiguous 8-elem
// chunks at k = kbase and k = kbase+8, where kbase = ks*32 + (lane&16 ? 16:0).
__device__ __forceinline__ v16bf load_afrag(const __bf16* rowp, int kbase)
{
    v8bf x0 = *reinterpret_cast<const v8bf*>(rowp + kbase);
    v8bf x1 = *reinterpret_cast<const v8bf*>(rowp + kbase + 8);
    v16bf f;
#pragma unroll
    for (int t = 0; t < 8; ++t) { f[t] = x0[t]; f[8 + t] = x1[t]; }
    return f;
}

// B 32x16 bf16 from transposed weights WT[n][k]: lane holds col n=lane&15,
// elements k = kbase..kbase+15 contiguous, kbase = ks*32 + (lane&16 ? 16:0).
__device__ __forceinline__ v16bf load_bfragT(const __bf16* wt, int n, int ldk, int kbase)
{
    const __bf16* p = wt + (size_t)n * ldk + kbase;
    v8bf x0 = *reinterpret_cast<const v8bf*>(p);
    v8bf x1 = *reinterpret_cast<const v8bf*>(p + 8);
    v16bf f;
#pragma unroll
    for (int t = 0; t < 8; ++t) { f[t] = x0[t]; f[8 + t] = x1[t]; }
    return f;
}

__device__ __forceinline__ v8f vzero8()
{
    v8f z;
#pragma unroll
    for (int r = 0; r < 8; ++r) z[r] = 0.f;
    return z;
}

__device__ __forceinline__ v8f wmma_bf16(v16bf a, v16bf b, v8f c)
{
    return __builtin_amdgcn_wmma_f32_16x16x32_bf16(false, a, false, b, (short)0, c, false, false);
}

// ---------------------------------------------------------------------------
// Pass 1: weights f32 row-major [K][N]  ->  bf16 transposed [N][Kpad] in d_ws
// ---------------------------------------------------------------------------
__global__ void convert_weights(const float* __restrict__ ew1, const float* __restrict__ ew2,
                                const float* __restrict__ ew3, const float* __restrict__ nw1,
                                const float* __restrict__ nw2, const float* __restrict__ nw3,
                                __bf16* __restrict__ ws)
{
    const unsigned stride = gridDim.x * blockDim.x;
    for (unsigned idx = blockIdx.x * blockDim.x + threadIdx.x; idx < WS_TOTAL; idx += stride) {
        float v;
        if (idx < EW2T_OFF) {                      // ew1t [512][256]
            unsigned r = idx, n = r >> 8, k = r & 255u;
            v = ew1[(size_t)k * HDIM + n];
        } else if (idx < EW3T_OFF) {               // ew2t [512][512]
            unsigned r = idx - EW2T_OFF, n = r >> 9, k = r & 511u;
            v = ew2[(size_t)k * HDIM + n];
        } else if (idx < NW1T_OFF) {               // ew3t [512][512]
            unsigned r = idx - EW3T_OFF, n = r >> 9, k = r & 511u;
            v = ew3[(size_t)k * HDIM + n];
        } else if (idx < NW2T_OFF) {               // nw1t [512][672]
            unsigned r = idx - NW1T_OFF, n = r / NAK, k = r % NAK;
            v = (k < DDIM + ADIM + HDIM) ? nw1[(size_t)k * HDIM + n] : 0.f;
        } else if (idx < NW3T_OFF) {               // nw2t [512][512]
            unsigned r = idx - NW2T_OFF, n = r >> 9, k = r & 511u;
            v = nw2[(size_t)k * HDIM + n];
        } else {                                   // nw3t [128][512]
            unsigned r = idx - NW3T_OFF, n = r >> 9, k = r & 511u;
            v = nw3[(size_t)k * DDIM + n];
        }
        ws[idx] = (__bf16)v;
    }
}

// ---------------------------------------------------------------------------
// Pass 2: fused GNN
// ---------------------------------------------------------------------------
__global__ __launch_bounds__(NTHREADS, 1)
void ctm_fused(const float* __restrict__ states, const int* __restrict__ action,
               const __bf16* __restrict__ ws,
               const float* __restrict__ eb1, const float* __restrict__ eb2,
               const float* __restrict__ elng, const float* __restrict__ elnb,
               const float* __restrict__ eb3,
               const float* __restrict__ nb1, const float* __restrict__ nb2,
               const float* __restrict__ nlng, const float* __restrict__ nlnb,
               const float* __restrict__ nb3,
               float* __restrict__ out)
{
    __shared__ __align__(16) __bf16 node_bf[KOBJ][DDIM];   // 4 KB
    __shared__ __align__(16) __bf16 buf1[CH][LDSTR];       // 50.7 KB
    __shared__ __align__(16) __bf16 buf2[CH][LDSTR];       // 50.7 KB
    __shared__ __align__(16) float  agg[KOBJ][HDIM];       // 32 KB
    __shared__ float red_sum[CH], red_sq[CH], ln_mu[CH], ln_iv[CH];

    const __bf16* ew1t = ws + EW1T_OFF;
    const __bf16* ew2t = ws + EW2T_OFF;
    const __bf16* ew3t = ws + EW3T_OFF;
    const __bf16* nw1t = ws + NW1T_OFF;
    const __bf16* nw2t = ws + NW2T_OFF;
    const __bf16* nw3t = ws + NW3T_OFF;

    const int b    = blockIdx.x;
    const int tid  = threadIdx.x;
    const int wave = tid >> 5;
    const int lane = tid & 31;
    const int mrow = lane & 15;
    const int hi    = (lane & 16) ? 8 : 0;    // D-frag row offset
    const int koffA = (lane & 16) ? 16 : 0;   // A-frag k offset
    const int koffB = (lane & 16) ? 16 : 0;   // B-frag k offset

    if (tid == 0) {    // warm L2 (global_prefetch_b8)
        __builtin_prefetch(ew1t, 0, 3); __builtin_prefetch(ew2t, 0, 3);
        __builtin_prefetch(ew3t, 0, 3); __builtin_prefetch(nw1t, 0, 3);
        __builtin_prefetch(nw2t, 0, 3); __builtin_prefetch(nw3t, 0, 3);
    }

    // ---- stage 0 -------------------------------------------------------------
    for (int idx = tid; idx < KOBJ * DDIM; idx += NTHREADS)
        node_bf[idx >> 7][idx & 127] = (__bf16)states[(size_t)b * (KOBJ * DDIM) + idx];
    for (int idx = tid; idx < KOBJ * HDIM; idx += NTHREADS)
        agg[idx >> 9][idx & 511] = 0.f;
    const int act = action[b];
    __syncthreads();

    // ================= edge MLP, 5 chunks of 48 edge rows ======================
    for (int c = 0; c < NCH; ++c) {
        const int ebase0 = c * CH;

        // per-chunk gather row pointers (lane holds row mrow of each M-tile)
        const __bf16 *rlo0, *rhi0, *rlo1, *rhi1, *rlo2, *rhi2;
        {
            int e0 = ebase0 + mrow, i0 = e0 / 15, t0 = e0 % 15, j0 = t0 + (t0 >= i0);
            int e1 = e0 + 16,       i1 = e1 / 15, t1 = e1 % 15, j1 = t1 + (t1 >= i1);
            int e2 = e0 + 32,       i2 = e2 / 15, t2 = e2 % 15, j2 = t2 + (t2 >= i2);
            rlo0 = node_bf[i0]; rhi0 = node_bf[j0];
            rlo1 = node_bf[i1]; rhi1 = node_bf[j1];
            rlo2 = node_bf[i2]; rhi2 = node_bf[j2];
        }

        // ---- GEMM1: concat(node_i,node_j)[48x256] @ ew1 -> relu -> buf1 ------
        for (int ntl = 0; ntl < 4; ++ntl) {
            const int nt = (wave << 2) + ntl;
            const int n  = (nt << 4) + mrow;
            v8f a0 = vzero8(), a1 = vzero8(), a2 = vzero8();
            for (int ks = 0; ks < 8; ++ks) {
                const v16bf bf = load_bfragT(ew1t, n, 2 * DDIM, (ks << 5) + koffB);
                const int  kb  = ((ks & 3) << 5) + koffA;
                const bool lo  = ks < 4;
                a0 = wmma_bf16(load_afrag(lo ? rlo0 : rhi0, kb), bf, a0);
                a1 = wmma_bf16(load_afrag(lo ? rlo1 : rhi1, kb), bf, a1);
                a2 = wmma_bf16(load_afrag(lo ? rlo2 : rhi2, kb), bf, a2);
            }
            const float bias = eb1[n];
#pragma unroll
            for (int r = 0; r < 8; ++r) {
                buf1[hi + r][n]      = (__bf16)fmaxf(a0[r] + bias, 0.f);
                buf1[16 + hi + r][n] = (__bf16)fmaxf(a1[r] + bias, 0.f);
                buf1[32 + hi + r][n] = (__bf16)fmaxf(a2[r] + bias, 0.f);
            }
        }
        for (int idx = tid; idx < CH; idx += NTHREADS) { red_sum[idx] = 0.f; red_sq[idx] = 0.f; }
        __syncthreads();

        // ---- GEMM2: buf1 @ ew2 -> raw+bias -> buf2, LN stats -----------------
        {
            const __bf16* ar0 = buf1[mrow];
            const __bf16* ar1 = buf1[16 + mrow];
            const __bf16* ar2 = buf1[32 + mrow];
            for (int ntl = 0; ntl < 4; ++ntl) {
                const int nt = (wave << 2) + ntl;
                const int n  = (nt << 4) + mrow;
                v8f a0 = vzero8(), a1 = vzero8(), a2 = vzero8();
                for (int ks = 0; ks < 16; ++ks) {
                    const v16bf bf = load_bfragT(ew2t, n, HDIM, (ks << 5) + koffB);
                    const int  kb  = (ks << 5) + koffA;
                    a0 = wmma_bf16(load_afrag(ar0, kb), bf, a0);
                    a1 = wmma_bf16(load_afrag(ar1, kb), bf, a1);
                    a2 = wmma_bf16(load_afrag(ar2, kb), bf, a2);
                }
                const float bias = eb2[n];
#pragma unroll
                for (int r = 0; r < 8; ++r) {
                    float x0 = a0[r] + bias, x1 = a1[r] + bias, x2 = a2[r] + bias;
                    buf2[hi + r][n]      = (__bf16)x0;
                    buf2[16 + hi + r][n] = (__bf16)x1;
                    buf2[32 + hi + r][n] = (__bf16)x2;
                    atomicAdd(&red_sum[hi + r],      x0); atomicAdd(&red_sq[hi + r],      x0 * x0);
                    atomicAdd(&red_sum[16 + hi + r], x1); atomicAdd(&red_sq[16 + hi + r], x1 * x1);
                    atomicAdd(&red_sum[32 + hi + r], x2); atomicAdd(&red_sq[32 + hi + r], x2 * x2);
                }
            }
        }
        __syncthreads();
        for (int idx = tid; idx < CH; idx += NTHREADS) {
            const float mu  = red_sum[idx] * (1.f / HDIM);
            const float var = red_sq[idx] * (1.f / HDIM) - mu * mu;
            ln_mu[idx] = mu;
            ln_iv[idx] = rsqrtf(var + LN_EPS);
        }
        __syncthreads();

        // ---- LN+ReLU applied once: buf2 -> buf1 ------------------------------
        for (int idx = tid; idx < CH * HDIM; idx += NTHREADS) {
            const int m = idx >> 9, k = idx & 511;
            const float x = (float)buf2[m][k];
            const float y = (x - ln_mu[m]) * ln_iv[m] * elng[k] + elnb[k];
            buf1[m][k] = (__bf16)fmaxf(y, 0.f);
        }
        __syncthreads();

        // ---- GEMM3: buf1 @ ew3 -> +bias -> scatter into agg ------------------
        {
            const __bf16* ar0 = buf1[mrow];
            const __bf16* ar1 = buf1[16 + mrow];
            const __bf16* ar2 = buf1[32 + mrow];
            for (int ntl = 0; ntl < 4; ++ntl) {
                const int nt = (wave << 2) + ntl;
                const int n  = (nt << 4) + mrow;
                v8f a0 = vzero8(), a1 = vzero8(), a2 = vzero8();
                for (int ks = 0; ks < 16; ++ks) {
                    const v16bf bf = load_bfragT(ew3t, n, HDIM, (ks << 5) + koffB);
                    const int  kb  = (ks << 5) + koffA;
                    a0 = wmma_bf16(load_afrag(ar0, kb), bf, a0);
                    a1 = wmma_bf16(load_afrag(ar1, kb), bf, a1);
                    a2 = wmma_bf16(load_afrag(ar2, kb), bf, a2);
                }
                const float bias = eb3[n];
#pragma unroll
                for (int r = 0; r < 8; ++r) {
                    const int e0 = ebase0 + hi + r;
                    atomicAdd(&agg[e0 / 15][n],        a0[r] + bias);
                    atomicAdd(&agg[(e0 + 16) / 15][n], a1[r] + bias);
                    atomicAdd(&agg[(e0 + 32) / 15][n], a2[r] + bias);
                }
            }
        }
        __syncthreads();
    }

    // ================= node MLP ================================================
    // materialize node input A = [16 x 672] (node | one-hot(action) | agg | 0)
    __bf16* nodeA = &buf1[0][0];          // flat reuse, stride NASTR
    for (int idx = tid; idx < KOBJ * NAK; idx += NTHREADS) {
        const int m = idx / NAK, k = idx % NAK;
        float v;
        if (k < DDIM)              v = (float)node_bf[m][k];
        else if (k < DDIM + ADIM)  v = (act == m * ADIM + (k - DDIM)) ? 1.f : 0.f;
        else if (k < DDIM + ADIM + HDIM) v = agg[m][k - (DDIM + ADIM)];
        else                       v = 0.f;
        nodeA[m * NASTR + k] = (__bf16)v;
    }
    __syncthreads();

    // node GEMM1: nodeA @ nw1 -> relu -> buf2 rows 0..15
    {
        const __bf16* nar = nodeA + mrow * NASTR;
        for (int ntl = 0; ntl < 4; ++ntl) {
            const int nt = (wave << 2) + ntl;
            const int n  = (nt << 4) + mrow;
            v8f acc = vzero8();
            for (int ks = 0; ks < 21; ++ks)
                acc = wmma_bf16(load_afrag(nar, (ks << 5) + koffA),
                                load_bfragT(nw1t, n, NAK, (ks << 5) + koffB), acc);
            const float bias = nb1[n];
#pragma unroll
            for (int r = 0; r < 8; ++r)
                buf2[hi + r][n] = (__bf16)fmaxf(acc[r] + bias, 0.f);
        }
    }
    for (int idx = tid; idx < KOBJ; idx += NTHREADS) { red_sum[idx] = 0.f; red_sq[idx] = 0.f; }
    __syncthreads();

    // node GEMM2: buf2 @ nw2 -> raw+bias -> buf1 rows 0..15, LN stats
    {
        const __bf16* nar = buf2[mrow];
        for (int ntl = 0; ntl < 4; ++ntl) {
            const int nt = (wave << 2) + ntl;
            const int n  = (nt << 4) + mrow;
            v8f acc = vzero8();
            for (int ks = 0; ks < 16; ++ks)
                acc = wmma_bf16(load_afrag(nar, (ks << 5) + koffA),
                                load_bfragT(nw2t, n, HDIM, (ks << 5) + koffB), acc);
            const float bias = nb2[n];
#pragma unroll
            for (int r = 0; r < 8; ++r) {
                const float x = acc[r] + bias;
                buf1[hi + r][n] = (__bf16)x;
                atomicAdd(&red_sum[hi + r], x);
                atomicAdd(&red_sq[hi + r], x * x);
            }
        }
    }
    __syncthreads();
    for (int idx = tid; idx < KOBJ; idx += NTHREADS) {
        const float mu  = red_sum[idx] * (1.f / HDIM);
        const float var = red_sq[idx] * (1.f / HDIM) - mu * mu;
        ln_mu[idx] = mu;
        ln_iv[idx] = rsqrtf(var + LN_EPS);
    }
    __syncthreads();
    for (int idx = tid; idx < KOBJ * HDIM; idx += NTHREADS) {
        const int m = idx >> 9, k = idx & 511;
        const float x = (float)buf1[m][k];
        const float y = (x - ln_mu[m]) * ln_iv[m] * nlng[k] + nlnb[k];
        buf2[m][k] = (__bf16)fmaxf(y, 0.f);
    }
    __syncthreads();

    // node GEMM3: buf2 @ nw3 [512x128] -> out (one 16-col tile per wave)
    {
        const __bf16* nar = buf2[mrow];
        const int n = (wave << 4) + mrow;
        v8f acc = vzero8();
        for (int ks = 0; ks < 16; ++ks)
            acc = wmma_bf16(load_afrag(nar, (ks << 5) + koffA),
                            load_bfragT(nw3t, n, HDIM, (ks << 5) + koffB), acc);
        const float bias = nb3[n];
#pragma unroll
        for (int r = 0; r < 8; ++r)
            out[((size_t)b * KOBJ + hi + r) * DDIM + n] = acc[r] + bias;
    }
}

extern "C" void kernel_launch(void* const* d_in, const int* in_sizes, int n_in,
                              void* d_out, int out_size, void* d_ws, size_t ws_size,
                              hipStream_t stream)
{
    (void)in_sizes; (void)n_in; (void)ws_size; (void)out_size;
    const float* states = (const float*)d_in[0];
    const int*   action = (const int*)d_in[1];
    // d_in[2] = edge_index: deterministic fully-connected structure, derived analytically.
    const float* ew1  = (const float*)d_in[3];
    const float* eb1  = (const float*)d_in[4];
    const float* ew2  = (const float*)d_in[5];
    const float* eb2  = (const float*)d_in[6];
    const float* elng = (const float*)d_in[7];
    const float* elnb = (const float*)d_in[8];
    const float* ew3  = (const float*)d_in[9];
    const float* eb3  = (const float*)d_in[10];
    const float* nw1  = (const float*)d_in[11];
    const float* nb1  = (const float*)d_in[12];
    const float* nw2  = (const float*)d_in[13];
    const float* nb2  = (const float*)d_in[14];
    const float* nlng = (const float*)d_in[15];
    const float* nlnb = (const float*)d_in[16];
    const float* nw3  = (const float*)d_in[17];
    const float* nb3  = (const float*)d_in[18];
    float* out = (float*)d_out;
    __bf16* ws = (__bf16*)d_ws;

    convert_weights<<<1296, NTHREADS, 0, stream>>>(ew1, ew2, ew3, nw1, nw2, nw3, ws);
    ctm_fused<<<512, NTHREADS, 0, stream>>>(states, action, ws,
                                            eb1, eb2, elng, elnb, eb3,
                                            nb1, nb2, nlng, nlnb, nb3,
                                            out);
}
